// GMViT_mini_31525059953264
// MI455X (gfx1250) — compile-verified, matrix operand order checked by reading, not run on hardware
//
#include <hip/hip_runtime.h>
#include <hip/hip_bf16.h>
#include <math.h>

// ---------------------------------------------------------------------------
// GMViT-mini fused pipeline for gfx1250 (wave32, WMMA bf16 16x16x32, f32 acc)
// ---------------------------------------------------------------------------

typedef __attribute__((ext_vector_type(16))) __bf16 v16bf;
typedef __attribute__((ext_vector_type(8)))  float  v8f;

#define WAVES   8
#define THREADS 256
#define STRA    40                       // padded LDS stride (bf16 elems) for A tile
#define STRB    40                       // padded LDS stride (bf16 elems) for B tile
#define SMEM_A_BYTES (16 * STRA * 2)     // 1280

__device__ __forceinline__ unsigned short f2bf(float f) {
  unsigned u = __float_as_uint(f);
  u += 0x7FFFu + ((u >> 16) & 1u);       // round-to-nearest-even
  return (unsigned short)(u >> 16);
}
__device__ __forceinline__ unsigned pack2bf(float lo, float hi) {
  return (unsigned)f2bf(lo) | ((unsigned)f2bf(hi) << 16);
}
// float atomic max via signed/unsigned integer ordering trick
__device__ __forceinline__ void atomicMaxF(float* addr, float val) {
  if (val >= 0.f) atomicMax((int*)addr, __float_as_int(val));
  else            atomicMin((unsigned int*)addr, __float_as_uint(val));
}

__global__ void init_neg_inf(float* p, int n) {
  int i = blockIdx.x * blockDim.x + threadIdx.x;
  if (i < n) p[i] = __int_as_float(0xFF800000);   // -inf
}

union bfrag { uint4 u[2]; v16bf v; };

// EPI: 0=bias store, 1=LN+att+viewmax, 2=LN, 3=BN split(+relu copy),
//      4=BN+relu, 5=bias store with N guard (pred)
// NPAD: padded N, multiple of 128 so every wave owns NPAD/128 tiles (no guards)
template<int EPI, int NPAD, int NREAL>
__global__ __launch_bounds__(THREADS)
void gemm_rows(const float* __restrict__ A, int K,
               const float* __restrict__ Bw,
               const float* __restrict__ bias,
               float* __restrict__ out0,
               float* __restrict__ out1,
               float* __restrict__ out2,
               const float* __restrict__ g0,
               const float* __restrict__ b0,
               const float* __restrict__ wa,
               const float* __restrict__ bap)
{
  constexpr int NTILES = NPAD / 16;
  constexpr int TPW    = NTILES / WAVES;     // tiles per wave (1,2,4)
  constexpr int BREPS  = NPAD / 16;          // B staging passes
  constexpr int SMEM_B = NPAD * STRB * 2;

  __shared__ char smem[SMEM_A_BYTES + SMEM_B];
  unsigned short* lA    = (unsigned short*)smem;
  unsigned short* lB    = (unsigned short*)(smem + SMEM_A_BYTES);
  float*          ltile = (float*)(smem + SMEM_A_BYTES);   // LN epilogue reuses B

  const int tid  = threadIdx.x;
  const int wave = tid >> 5;
  const int lane = tid & 31;
  const int l15  = lane & 15;
  const int half = lane >> 4;
  const int rowbase = blockIdx.x * 16;

  v8f acc[TPW];
  #pragma unroll
  for (int j = 0; j < TPW; ++j)
    #pragma unroll
    for (int e = 0; e < 8; ++e) acc[j][e] = 0.f;

  // per-lane A fragment byte offsets in LDS (ISA 16-bit A 16x32 layout)
  const int aoff0 = (l15 * STRA + half * 8) * 2;
  // per-lane B fragment base: contiguous tiles per wave -> constant DS offsets
  const int boffbase = ((wave * TPW * 16 + l15) * STRB + half * 8) * 2;

  const int arow = tid >> 4;          // A staging: 16 rows x 32 k, 2 k per thread
  const int akp  = (tid & 15) * 2;

  for (int k0 = 0; k0 < K; k0 += 32) {
    __syncthreads();
    // ---- stage A tile (f32 -> bf16) ----
    {
      const float* ap = A + (size_t)(rowbase + arow) * K + (k0 + akp);
      float2 av = *(const float2*)ap;
      *(unsigned*)((char*)lA + arow * (STRA * 2) + akp * 2) = pack2bf(av.x, av.y);
      if (k0 + 32 < K)
        __builtin_prefetch(ap + 32, 0, 3);   // global_prefetch_b8, WGP scope
    }
    // ---- stage B tile transposed to [n][k] (f32 -> bf16) ----
    #pragma unroll
    for (int rep = 0; rep < BREPS; ++rep) {
      int p  = rep * THREADS + tid;
      int n  = p & (NPAD - 1);
      int kk = (p / NPAD) * 2;
      float blo = 0.f, bhi = 0.f;
      if (NREAL == NPAD || n < NREAL) {
        blo = Bw[(size_t)(k0 + kk)     * NREAL + n];
        bhi = Bw[(size_t)(k0 + kk + 1) * NREAL + n];
      }
      *(unsigned*)((char*)lB + n * (STRB * 2) + kk * 2) = pack2bf(blo, bhi);
    }
    __syncthreads();

    // ---- fragments: rotated loop so tile j+1 loads issue under WMMA j ----
    bfrag af;
    af.u[0] = *(const uint4*)((char*)lA + aoff0);
    af.u[1] = *(const uint4*)((char*)lA + aoff0 + 32);

    bfrag cur;
    cur.u[0] = *(const uint4*)((char*)lB + boffbase);
    cur.u[1] = *(const uint4*)((char*)lB + boffbase + 32);
    #pragma unroll
    for (int j = 0; j < TPW; ++j) {
      bfrag nxt;
      if (j + 1 < TPW) {
        int bo = boffbase + (j + 1) * (16 * STRB * 2);
        nxt.u[0] = *(const uint4*)((char*)lB + bo);
        nxt.u[1] = *(const uint4*)((char*)lB + bo + 32);
      }
      acc[j] = __builtin_amdgcn_wmma_f32_16x16x32_bf16(
          false, af.v, false, cur.v, (short)0, acc[j], false, false);
      if (j + 1 < TPW) cur = nxt;
    }
  }

  // ----------------------------- epilogues ------------------------------
  if (EPI == 1 || EPI == 2) {
    __syncthreads();                       // done reading lB as bf16
    constexpr int TSTR = NPAD + 4;
    #pragma unroll
    for (int j = 0; j < TPW; ++j) {
      int col = (wave * TPW + j) * 16 + l15;
      #pragma unroll
      for (int r = 0; r < 8; ++r) {
        int row = r + 8 * half;            // C/D layout: VGPR r -> M = r + 8*half
        ltile[row * TSTR + col] = acc[j][r] + bias[col];
      }
    }
    __syncthreads();

    int row = tid >> 4;
    int idx = tid & 15;
    float sum = 0.f, ssq = 0.f;
    #pragma unroll
    for (int c0 = 0; c0 < NPAD; c0 += 16) {
      float v = ltile[row * TSTR + c0 + idx];
      sum += v; ssq += v * v;
    }
    #pragma unroll
    for (int m = 1; m < 16; m <<= 1) {
      sum += __shfl_xor(sum, m, 16);
      ssq += __shfl_xor(ssq, m, 16);
    }
    constexpr float invN = 1.f / (float)NPAD;
    float mean = sum * invN;
    float var  = ssq * invN - mean * mean;
    float rstd = rsqrtf(var + 1e-5f);
    int grow = rowbase + row;

    float pa = 0.f;
    #pragma unroll
    for (int c0 = 0; c0 < NPAD; c0 += 16) {
      int c = c0 + idx;
      float x = ltile[row * TSTR + c];
      float y = (x - mean) * rstd * g0[c] + b0[c];
      out0[(size_t)grow * NPAD + c] = y;
      if (EPI == 1) {
        pa += y * wa[c];
        atomicMaxF(&out2[(size_t)(grow / 20) * NPAD + c], x);  // max over views of x1
      }
    }
    if (EPI == 1) {
      #pragma unroll
      for (int m = 1; m < 16; m <<= 1) pa += __shfl_xor(pa, m, 16);
      if (idx == 0)
        out1[grow] = 1.f / (1.f + __expf(-(pa + bap[0])));
    }
  } else {
    const float BNS = 0.9999950000374997f;   // 1/sqrt(1 + 1e-5)
    #pragma unroll
    for (int j = 0; j < TPW; ++j) {
      int col = (wave * TPW + j) * 16 + l15;
      float bx = (NREAL == NPAD || col < NREAL) ? bias[col] : 0.f;
      float gv = 0.f, bv = 0.f;
      if (EPI == 3 || EPI == 4) { gv = g0[col]; bv = b0[col]; }
      #pragma unroll
      for (int r = 0; r < 8; ++r) {
        int row = r + 8 * half;
        size_t grow = (size_t)(rowbase + row);
        float x = acc[j][r] + bx;
        if (EPI == 0) {
          out0[grow * NPAD + col] = x;
        } else if (EPI == 3) {
          float y = x * BNS * gv + bv;
          out0[grow * NPAD + col] = y;                // f_global
          out1[grow * NPAD + col] = fmaxf(y, 0.f);    // relu(h)
        } else if (EPI == 4) {
          out0[grow * NPAD + col] = fmaxf(x * BNS * gv + bv, 0.f);
        } else { // EPI == 5
          if (col < NREAL) out0[grow * NREAL + col] = x;
        }
      }
    }
  }
}

extern "C" void kernel_launch(void* const* d_in, const int* in_sizes, int n_in,
                              void* d_out, int out_size, void* d_ws, size_t ws_size,
                              hipStream_t stream) {
  const float* x    = (const float*)d_in[0];
  const float* We   = (const float*)d_in[1];
  const float* be   = (const float*)d_in[2];
  const float* W1   = (const float*)d_in[3];
  const float* b1   = (const float*)d_in[4];
  const float* Wa   = (const float*)d_in[5];
  const float* ba   = (const float*)d_in[6];
  const float* W2   = (const float*)d_in[7];
  const float* b2   = (const float*)d_in[8];
  const float* ln_g = (const float*)d_in[9];
  const float* ln_b = (const float*)d_in[10];
  const float* h1W  = (const float*)d_in[11];
  const float* h1b  = (const float*)d_in[12];
  const float* bn1g = (const float*)d_in[13];
  const float* bn1b = (const float*)d_in[14];
  const float* h2W  = (const float*)d_in[15];
  const float* h2b  = (const float*)d_in[16];
  const float* bn2g = (const float*)d_in[17];
  const float* bn2b = (const float*)d_in[18];
  const float* h3W  = (const float*)d_in[19];
  const float* h3b  = (const float*)d_in[20];

  const int BV = 81920, B = 4096, D = 512;

  float* out    = (float*)d_out;
  float* f_cnn  = out;                               // [81920,512]
  float* f_vit1 = f_cnn  + (size_t)BV * D;           // [81920,512]
  float* att    = f_vit1 + (size_t)BV * D;           // [4096,20]
  float* f_vit2 = att    + (size_t)B * 20;           // [4096,512]
  float* f_glob = f_vit2 + (size_t)B * D;            // [4096,512]
  float* pred   = f_glob + (size_t)B * D;            // [4096,40]

  float* vmax  = (float*)d_ws;                       // [4096,512] view-max of x1
  float* hrelu = vmax  + (size_t)B * D;              // [4096,512]
  float* h2    = hrelu + (size_t)B * D;              // [4096,256]

  init_neg_inf<<<(B * D + 255) / 256, 256, 0, stream>>>(vmax, B * D);

  dim3 blk(THREADS);
  // f_cnn = x @ We + be
  gemm_rows<0, 512, 512><<<BV / 16, blk, 0, stream>>>(x, D, We, be,
      f_cnn, nullptr, nullptr, nullptr, nullptr, nullptr, nullptr);
  // x1 = f_cnn @ W1 + b1 ; f_vit1 = LN(x1) ; att = sigmoid(f_vit1@Wa+ba) ; vmax = max_v x1
  gemm_rows<1, 512, 512><<<BV / 16, blk, 0, stream>>>(f_cnn, D, W1, b1,
      f_vit1, att, vmax, ln_g, ln_b, Wa, ba);
  // f_vit2 = LN(vmax @ W2 + b2)
  gemm_rows<2, 512, 512><<<B / 16, blk, 0, stream>>>(vmax, D, W2, b2,
      f_vit2, nullptr, nullptr, ln_g, ln_b, nullptr, nullptr);
  // f_global = BN(f_vit2 @ h1W + h1b) ; hrelu = relu(f_global)
  gemm_rows<3, 512, 512><<<B / 16, blk, 0, stream>>>(f_vit2, D, h1W, h1b,
      f_glob, hrelu, nullptr, bn1g, bn1b, nullptr, nullptr);
  // h2 = relu(BN(hrelu @ h2W + h2b))
  gemm_rows<4, 256, 256><<<B / 16, blk, 0, stream>>>(hrelu, D, h2W, h2b,
      h2, nullptr, nullptr, bn2g, bn2b, nullptr, nullptr);
  // pred = h2 @ h3W + h3b   (N=40 padded to 128 -> 1 tile per wave, no divergence)
  gemm_rows<5, 128, 40><<<B / 16, blk, 0, stream>>>(h2, 256, h3W, h3b,
      pred, nullptr, nullptr, nullptr, nullptr, nullptr, nullptr);
}
